// TaskSNN_30236569764515
// MI455X (gfx1250) — compile-verified, matrix operand order checked by reading, not run on hardware
//
#include <hip/hip_runtime.h>

typedef __attribute__((ext_vector_type(16))) _Float16 v16h;
typedef __attribute__((ext_vector_type(8)))  float    v8f;
typedef __attribute__((ext_vector_type(4)))  unsigned uv4;
typedef __attribute__((ext_vector_type(8)))  unsigned uv8;
typedef __attribute__((ext_vector_type(4)))  float    fv4;

union FragU { uv4 u[2]; v16h h; };

#define B_  512
#define T_  1000
#define C_  64
#define H_  256
#define O_  4
#define BM  16          // batch rows per workgroup

// packed-weight layout offsets in workspace (uint units)
#define WIN_U  0u       // [16][2][32][8]   (N16, K/32, lane, dword)
#define V1_U   8192u    // [16][8][32][8]
#define WH_U   40960u
#define V2_U   73728u
#define WS_UINTS 106496u

__device__ __forceinline__ v8f wmma16(const FragU& a, const FragU& b, v8f c) {
  return __builtin_amdgcn_wmma_f32_16x16x32_f16(false, a.h, false, b.h, (short)0, c, false, false);
}

// ---------------------------------------------------------------------------
// Pack fp32 weights [N][K] into f16 WMMA B-fragment layout:
//   element [n16][f][lane][j] = packed pair (W[n][k], W[n][k+1]),
//   n = n16*16 + (lane&15), kg = lane>>4,
//   k = f*32 + kg*8 + (j<4 ? 2j : 16+2(j-4))
// ---------------------------------------------------------------------------
__global__ void snn_pack_b_kernel(const float* __restrict__ Win,
                                  const float* __restrict__ V1,
                                  const float* __restrict__ Wh,
                                  const float* __restrict__ V2,
                                  unsigned* __restrict__ ws) {
  unsigned tid = blockIdx.x * blockDim.x + threadIdx.x;
  const float* W; unsigned K, base, loc;
  if (tid < 8192u)        { W = Win; K = 64u;  base = WIN_U; loc = tid; }
  else if (tid < 40960u)  { W = V1;  K = 256u; base = V1_U;  loc = tid - 8192u; }
  else if (tid < 73728u)  { W = Wh;  K = 256u; base = WH_U;  loc = tid - 40960u; }
  else if (tid < WS_UINTS){ W = V2;  K = 256u; base = V2_U;  loc = tid - 73728u; }
  else return;
  unsigned j = loc & 7u, l = (loc >> 3) & 31u, rest = loc >> 8;
  unsigned kf = K >> 5;                       // K/32 fragments
  unsigned f = rest % kf, n16 = rest / kf;
  unsigned n = n16 * 16u + (l & 15u), kg = l >> 4;
  unsigned k = f * 32u + kg * 8u + ((j < 4u) ? 2u * j : 16u + 2u * (j - 4u));
  union { _Float16 h[2]; unsigned u; } p;
  p.h[0] = (_Float16)W[n * K + k];
  p.h[1] = (_Float16)W[n * K + k + 1u];
  ws[base + loc] = p.u;
}

// ---------------------------------------------------------------------------
// TDM: async 2D tile load (BM rows x C_ cols of f32, row stride T_*C_) -> LDS
// ---------------------------------------------------------------------------
__device__ __forceinline__ unsigned rfl(unsigned v) {
  return (unsigned)__builtin_amdgcn_readfirstlane((int)v);
}

__device__ __forceinline__ void tdm_load_x(const float* x, unsigned b0, unsigned t,
                                           unsigned lds_byte_off) {
  unsigned long long ga = (unsigned long long)(size_t)x +
      (((unsigned long long)b0 * (unsigned long long)T_ + (unsigned long long)t)
        * (unsigned long long)C_) * 4ull;
  uv4 g0; uv8 g1;
  g0[0] = rfl(1u);                                   // count=1, is_restore=0
  g0[1] = rfl(lds_byte_off);                         // lds_addr
  g0[2] = rfl((unsigned)ga);                         // global_addr[31:0]
  g0[3] = rfl(((unsigned)(ga >> 32)) | 0x80000000u); // global_addr[56:32] | type=2
  g1[0] = rfl(2u << 16);                             // data_size = 4 bytes
  g1[1] = rfl(((unsigned)C_) << 16);                 // tensor_dim0 lo16 (=64)
  g1[2] = rfl(((unsigned)BM) << 16);                 // dim0 hi=0 | tensor_dim1 lo16 (=16)
  g1[3] = rfl(((unsigned)C_) << 16);                 // dim1 hi=0 | tile_dim0 (=64)
  g1[4] = rfl((unsigned)BM);                         // tile_dim1 (=16), tile_dim2=0
  g1[5] = rfl((unsigned)(T_ * C_));                  // tensor_dim0_stride lo32
  g1[6] = rfl(0u);                                   // stride hi | dim1_stride lo
  g1[7] = rfl(0u);
  asm volatile("tensor_load_to_lds %0, %1" :: "s"(g0), "s"(g1) : "memory");
}

// ---------------------------------------------------------------------------
// Persistent scan kernel: 32 blocks x 512 threads (16 waves). Wave w owns
// H-columns [16w, 16w+16). mem1/mem2 kept in WMMA C-layout registers.
// ---------------------------------------------------------------------------
__global__ __launch_bounds__(512, 1)
void snn_scan_kernel(const float* __restrict__ x,
                     const float* __restrict__ bn_g,  const float* __restrict__ bn_b,
                     const float* __restrict__ bn_mean, const float* __restrict__ bn_var,
                     const float* __restrict__ b_in,
                     const float* __restrict__ ln1_g, const float* __restrict__ ln1_b,
                     const float* __restrict__ V1_b,
                     const float* __restrict__ b_h,
                     const float* __restrict__ ln2_g, const float* __restrict__ ln2_b,
                     const float* __restrict__ V2_b,
                     const float* __restrict__ W_out, const float* __restrict__ b_out,
                     const unsigned* __restrict__ ws,
                     float* __restrict__ out) {
  __shared__ __align__(16) float    xbuf[2][BM][C_];   // TDM double buffer
  __shared__ __align__(16) float    act[BM][H_];       // f32 staging / layernorm
  __shared__ __align__(16) _Float16 A0[BM][C_];        // BN output (GEMM1 A)
  __shared__ __align__(16) _Float16 A2[BM][H_];        // res1 (GEMM-Wh A)
  __shared__ __align__(16) _Float16 spk1[BM][H_];
  __shared__ __align__(16) _Float16 spk2[BM][H_];
  __shared__ __align__(16) float    bnsc[C_], bnmn[C_], bnbs[C_];
  __shared__ __align__(16) float    woutl[O_][H_];

  const int lane = threadIdx.x & 31;
  const int wave = threadIdx.x >> 5;          // 0..15 == H n-tile
  const int rA   = lane & 15;                 // A-fragment row
  const int kg   = (lane >> 4) & 1;           // A/B-fragment K group
  const int col  = wave * 16 + rA;            // owned H column
  const unsigned b0 = blockIdx.x * BM;
  const uv4* ws4 = (const uv4*)ws;

  // ---- init: constants to LDS / registers, zero spike state ----
  for (int e = threadIdx.x; e < C_; e += 512) {
    bnsc[e] = bn_g[e] * rsqrtf(bn_var[e] + 1e-5f);
    bnmn[e] = bn_mean[e];
    bnbs[e] = bn_b[e];
  }
  for (int e = threadIdx.x; e < O_ * H_; e += 512) ((float*)woutl)[e] = W_out[e];
  for (int e = threadIdx.x; e < BM * H_; e += 512) {
    ((_Float16*)spk1)[e] = (_Float16)0.0f;
    ((_Float16*)spk2)[e] = (_Float16)0.0f;
  }
  const float bin_reg = b_in[col];
  const float v1b_reg = V1_b[col];
  const float bh_reg  = b_h[col];
  const float v2b_reg = V2_b[col];
  float g1r[8], b1r[8], g2r[8], b2r[8];
#pragma unroll
  for (int i = 0; i < 8; ++i) {
    int k = lane * 8 + i;
    g1r[i] = ln1_g[k]; b1r[i] = ln1_b[k];
    g2r[i] = ln2_g[k]; b2r[i] = ln2_b[k];
  }
  float mem1[8], mem2[8];
#pragma unroll
  for (int i = 0; i < 8; ++i) { mem1[i] = 0.0f; mem2[i] = 0.0f; }
  float mo = 0.0f, acc = 0.0f;
  const float bo_reg = (threadIdx.x < 64) ? b_out[threadIdx.x & 3] : 0.0f;

  // initial x tile (t=0) via TDM
  if (wave == 0) {
    tdm_load_x(x, b0, 0u, (unsigned)(size_t)&xbuf[0][0][0]);
    __builtin_amdgcn_s_wait_tensorcnt(0);
  }
  __syncthreads();

  int cb = 0;
  for (int t = 0; t < T_; ++t) {
    // -- phase 0: TDM prefetch t+1; BatchNorm -> A0 (f16) --
    if (wave == 0 && t + 1 < T_)
      tdm_load_x(x, b0, (unsigned)(t + 1), (unsigned)(size_t)&xbuf[cb ^ 1][0][0]);
    for (int e = threadIdx.x; e < BM * C_; e += 512) {
      int r = e >> 6, c = e & 63;
      A0[r][c] = (_Float16)((xbuf[cb][r][c] - bnmn[c]) * bnsc[c] + bnbs[c]);
    }
    __syncthreads();

    // -- phase 1: A0 @ W_in^T + b_in -> act --
    {
      v8f c = {0.f,0.f,0.f,0.f,0.f,0.f,0.f,0.f};
#pragma unroll
      for (int f = 0; f < 2; ++f) {
        FragU a, bf;
        const _Float16* ap = &A0[rA][f * 32 + kg * 8];
        a.u[0] = *(const uv4*)ap; a.u[1] = *(const uv4*)(ap + 16);
        unsigned i4 = (WIN_U >> 2) + ((((unsigned)wave * 2u + f) * 32u + lane) << 1);
        bf.u[0] = ws4[i4]; bf.u[1] = ws4[i4 + 1];
        c = wmma16(a, bf, c);
      }
#pragma unroll
      for (int i = 0; i < 8; ++i) {
        int m = (lane < 16) ? i : (8 + i);
        act[m][col] = c[i] + bin_reg;
      }
    }
    __syncthreads();

    // -- phase 2: LayerNorm1 in place (wave per row, shfl reduction) --
    {
      int r = wave;
      fv4 v0 = *(const fv4*)&act[r][lane * 8];
      fv4 v1 = *(const fv4*)&act[r][lane * 8 + 4];
      float s = v0[0]+v0[1]+v0[2]+v0[3]+v1[0]+v1[1]+v1[2]+v1[3];
      float q = v0[0]*v0[0]+v0[1]*v0[1]+v0[2]*v0[2]+v0[3]*v0[3]
              + v1[0]*v1[0]+v1[1]*v1[1]+v1[2]*v1[2]+v1[3]*v1[3];
#pragma unroll
      for (int off = 16; off > 0; off >>= 1) {
        s += __shfl_xor(s, off); q += __shfl_xor(q, off);
      }
      float mean = s * (1.0f / 256.0f);
      float var  = q * (1.0f / 256.0f) - mean * mean;
      float rs   = rsqrtf(var + 1e-5f);
      float vv[8] = {v0[0],v0[1],v0[2],v0[3],v1[0],v1[1],v1[2],v1[3]};
#pragma unroll
      for (int i = 0; i < 8; ++i)
        act[r][lane * 8 + i] = (vv[i] - mean) * rs * g1r[i] + b1r[i];
    }
    __syncthreads();

    // -- phase 3: spk1 @ V1^T; LIF-1 update; new spk1, res1 -> A2 --
    {
      FragU sa[8];
#pragma unroll
      for (int f = 0; f < 8; ++f) {
        const _Float16* ap = &spk1[rA][f * 32 + kg * 8];
        sa[f].u[0] = *(const uv4*)ap; sa[f].u[1] = *(const uv4*)(ap + 16);
      }
      __syncthreads();   // everyone has old spikes before overwrite
      v8f c = {0.f,0.f,0.f,0.f,0.f,0.f,0.f,0.f};
#pragma unroll
      for (int f = 0; f < 8; ++f) {
        FragU bf;
        unsigned i4 = (V1_U >> 2) + ((((unsigned)wave * 8u + f) * 32u + lane) << 1);
        bf.u[0] = ws4[i4]; bf.u[1] = ws4[i4 + 1];
        c = wmma16(sa[f], bf, c);
      }
#pragma unroll
      for (int i = 0; i < 8; ++i) {
        int m = (lane < 16) ? i : (8 + i);
        float curin = act[m][col];
        float mm = mem1[i];
        float rst = (mm > 1.0f) ? 1.0f : 0.0f;
        mm = 0.9f * mm + curin + c[i] + v1b_reg - rst;
        mem1[i] = mm;
        float sp = (mm - 1.0f > 0.0f) ? 1.0f : 0.0f;
        spk1[m][col] = (_Float16)sp;
        A2[m][col]   = (_Float16)(sp + curin);
      }
    }
    __syncthreads();

    // -- phase 4: res1 @ W_h^T + b_h -> act --
    {
      v8f c = {0.f,0.f,0.f,0.f,0.f,0.f,0.f,0.f};
#pragma unroll
      for (int f = 0; f < 8; ++f) {
        FragU a, bf;
        const _Float16* ap = &A2[rA][f * 32 + kg * 8];
        a.u[0] = *(const uv4*)ap; a.u[1] = *(const uv4*)(ap + 16);
        unsigned i4 = (WH_U >> 2) + ((((unsigned)wave * 8u + f) * 32u + lane) << 1);
        bf.u[0] = ws4[i4]; bf.u[1] = ws4[i4 + 1];
        c = wmma16(a, bf, c);
      }
#pragma unroll
      for (int i = 0; i < 8; ++i) {
        int m = (lane < 16) ? i : (8 + i);
        act[m][col] = c[i] + bh_reg;
      }
    }
    __syncthreads();

    // -- phase 4b: LayerNorm2 in place --
    {
      int r = wave;
      fv4 v0 = *(const fv4*)&act[r][lane * 8];
      fv4 v1 = *(const fv4*)&act[r][lane * 8 + 4];
      float s = v0[0]+v0[1]+v0[2]+v0[3]+v1[0]+v1[1]+v1[2]+v1[3];
      float q = v0[0]*v0[0]+v0[1]*v0[1]+v0[2]*v0[2]+v0[3]*v0[3]
              + v1[0]*v1[0]+v1[1]*v1[1]+v1[2]*v1[2]+v1[3]*v1[3];
#pragma unroll
      for (int off = 16; off > 0; off >>= 1) {
        s += __shfl_xor(s, off); q += __shfl_xor(q, off);
      }
      float mean = s * (1.0f / 256.0f);
      float var  = q * (1.0f / 256.0f) - mean * mean;
      float rs   = rsqrtf(var + 1e-5f);
      float vv[8] = {v0[0],v0[1],v0[2],v0[3],v1[0],v1[1],v1[2],v1[3]};
#pragma unroll
      for (int i = 0; i < 8; ++i)
        act[r][lane * 8 + i] = (vv[i] - mean) * rs * g2r[i] + b2r[i];
    }
    __syncthreads();

    // -- phase 5: spk2 @ V2^T; LIF-2 update; res2 (f32) -> act --
    {
      FragU sa[8];
#pragma unroll
      for (int f = 0; f < 8; ++f) {
        const _Float16* ap = &spk2[rA][f * 32 + kg * 8];
        sa[f].u[0] = *(const uv4*)ap; sa[f].u[1] = *(const uv4*)(ap + 16);
      }
      __syncthreads();
      v8f c = {0.f,0.f,0.f,0.f,0.f,0.f,0.f,0.f};
#pragma unroll
      for (int f = 0; f < 8; ++f) {
        FragU bf;
        unsigned i4 = (V2_U >> 2) + ((((unsigned)wave * 8u + f) * 32u + lane) << 1);
        bf.u[0] = ws4[i4]; bf.u[1] = ws4[i4 + 1];
        c = wmma16(sa[f], bf, c);
      }
#pragma unroll
      for (int i = 0; i < 8; ++i) {
        int m = (lane < 16) ? i : (8 + i);
        float cur2 = act[m][col];
        float mm = mem2[i];
        float rst = (mm > 1.0f) ? 1.0f : 0.0f;
        mm = 0.9f * mm + cur2 + c[i] + v2b_reg - rst;
        mem2[i] = mm;
        float sp = (mm - 1.0f > 0.0f) ? 1.0f : 0.0f;
        spk2[m][col] = (_Float16)sp;
        act[m][col]  = sp + (float)A2[m][col];   // res2
      }
    }
    __syncthreads();

    // -- phase 6: output projection + leaky mem_out accumulation --
    if (threadIdx.x < 64) {
      int r = threadIdx.x >> 2, o = threadIdx.x & 3;
      float s = 0.0f;
#pragma unroll 8
      for (int k = 0; k < H_; ++k) s += act[r][k] * woutl[o][k];
      mo = 0.9f * mo + s + bo_reg;
      acc += mo;
    }
    if (wave == 0) __builtin_amdgcn_s_wait_tensorcnt(0);
    __syncthreads();   // next x tile + all LDS phases settled
    cb ^= 1;
  }

  if (threadIdx.x < 64) {
    int r = threadIdx.x >> 2, o = threadIdx.x & 3;
    out[(b0 + r) * O_ + o] = acc * (1.0f / (float)T_);
  }
}

extern "C" void kernel_launch(void* const* d_in, const int* in_sizes, int n_in,
                              void* d_out, int out_size, void* d_ws, size_t ws_size,
                              hipStream_t stream) {
  (void)in_sizes; (void)n_in; (void)out_size; (void)ws_size;
  unsigned* ws = (unsigned*)d_ws;
  snn_pack_b_kernel<<<(WS_UINTS + 255u) / 256u, 256, 0, stream>>>(
      (const float*)d_in[5],  // W_in
      (const float*)d_in[9],  // V1_W
      (const float*)d_in[11], // W_h
      (const float*)d_in[15], // V2_W
      ws);
  snn_scan_kernel<<<B_ / BM / 1, 512, 0, stream>>>(   // 32 blocks
      (const float*)d_in[0],
      (const float*)d_in[1],  (const float*)d_in[2],
      (const float*)d_in[3],  (const float*)d_in[4],
      (const float*)d_in[6],
      (const float*)d_in[7],  (const float*)d_in[8],
      (const float*)d_in[10],
      (const float*)d_in[12],
      (const float*)d_in[13], (const float*)d_in[14],
      (const float*)d_in[16],
      (const float*)d_in[17], (const float*)d_in[18],
      (const unsigned*)d_ws,
      (float*)d_out);
}
